// Transformer_80977313398936
// MI455X (gfx1250) — matrix operand order repaired and checked by reading, not compiled, not run on hardware
//
#include <hip/hip_runtime.h>

// ---------------------------------------------------------------------------
// Types for CDNA5 WMMA (wave32, 16x16x32 bf16 -> f32)
// ---------------------------------------------------------------------------
typedef __attribute__((ext_vector_type(16))) __bf16 v16bf;
typedef __attribute__((ext_vector_type(8)))  float  v8f;

union BF16Frag { v16bf v; unsigned int u[8]; };
union F32Frag  { v8f   v; float        f[8]; };

__device__ __forceinline__ unsigned int f2bf(float f) {
  unsigned int u = __float_as_uint(f);
  return (u + 0x7FFFu + ((u >> 16) & 1u)) >> 16;  // round-to-nearest-even
}
__device__ __forceinline__ unsigned int pack2bf(float a, float b) {
  return f2bf(a) | (f2bf(b) << 16);
}

#define WMMA_BF16(A_, B_, C_)                                                  \
  __builtin_amdgcn_wmma_f32_16x16x32_bf16(false, (A_), false, (B_), (short)0, \
                                          (C_), false, false)

// Model constants
#define MD    1024   // model dim
#define NH    16     // heads
#define HD    64     // head dim
#define SEQ   2048
#define BATCH 2
#define ROWS  (BATCH * SEQ)  // 4096
#define FFN   4096

// ---------------------------------------------------------------------------
// LayerNorm: one block (256 thr) per row of 1024
// ---------------------------------------------------------------------------
__global__ __launch_bounds__(256) void ln_kernel(const float* __restrict__ x,
                                                 const float* __restrict__ g,
                                                 const float* __restrict__ bt,
                                                 float* __restrict__ out) {
  __shared__ float ssum[8], ssq[8];
  int row = blockIdx.x;
  const float* xr = x + (size_t)row * MD;
  int c = threadIdx.x * 4;
  float4 v = *(const float4*)(xr + c);
  float s  = v.x + v.y + v.z + v.w;
  float s2 = v.x * v.x + v.y * v.y + v.z * v.z + v.w * v.w;
#pragma unroll
  for (int o = 16; o > 0; o >>= 1) {
    s  += __shfl_xor(s, o);
    s2 += __shfl_xor(s2, o);
  }
  if ((threadIdx.x & 31) == 0) { ssum[threadIdx.x >> 5] = s; ssq[threadIdx.x >> 5] = s2; }
  __syncthreads();
  s = 0.f; s2 = 0.f;
#pragma unroll
  for (int i = 0; i < 8; ++i) { s += ssum[i]; s2 += ssq[i]; }
  float mu   = s * (1.f / MD);
  float var  = s2 * (1.f / MD) - mu * mu;
  float rstd = rsqrtf(var + 1e-3f);
  float4 gg = *(const float4*)(g + c);
  float4 bb = *(const float4*)(bt + c);
  float4 o;
  o.x = (v.x - mu) * rstd * gg.x + bb.x;
  o.y = (v.y - mu) * rstd * gg.y + bb.y;
  o.z = (v.z - mu) * rstd * gg.z + bb.z;
  o.w = (v.w - mu) * rstd * gg.w + bb.w;
  *(float4*)(out + (size_t)row * MD + c) = o;
}

// ---------------------------------------------------------------------------
// bf16-WMMA GEMM: C[M,N] = act(A[M,K] @ B[K,N] + bias) (+ residual)
// Block tile 128x256, BK=32, 256 threads = 8 waves (2 M x 4 N), wave tile
// 64x64 (16 WMMA per K-step). Double-buffered LDS (one barrier / K-step),
// f32 operands converted to bf16 (RNE) while staging.
// ACT: 0 = none, 1 = softplus
// ---------------------------------------------------------------------------
template <int ACT, bool HAS_BIAS, bool HAS_RES>
__global__ __launch_bounds__(256) void gemm_kernel(
    const float* __restrict__ A, const float* __restrict__ Bm,
    const float* __restrict__ bias, const float* __restrict__ res,
    float* __restrict__ C, int M, int N, int K) {
  __shared__ unsigned short As[2][128][36];  // [m][k] bf16
  __shared__ unsigned short Bs[2][256][36];  // [n][k] bf16 (K-innermost)

  int tid  = threadIdx.x;
  int lane = tid & 31, wid = tid >> 5;
  int hi   = (lane >> 4) & 1, l16 = lane & 15;
  int m0   = blockIdx.y * 128, n0 = blockIdx.x * 256;
  int waveM = wid >> 2, waveN = wid & 3;  // 2 x 4 waves, each 64x64

  F32Frag acc[4][4];
#pragma unroll
  for (int mi = 0; mi < 4; ++mi)
#pragma unroll
    for (int ni = 0; ni < 4; ++ni)
#pragma unroll
      for (int r = 0; r < 8; ++r) acc[mi][ni].f[r] = 0.f;

  // staging coordinates
  int arow = tid >> 1, acol = (tid & 1) * 16;        // A: 16 f32 / thread
  int bk4  = (tid >> 5) * 4, bcol8 = (tid & 31) * 8; // B: 4k x 8n / thread

  const float* apg = A + (size_t)(m0 + arow) * K + acol;
  const float* bpg = Bm + (size_t)bk4 * N + n0 + bcol8;

  int nk = K / 32;

  // -------- stage one 128x32 A tile + 32x256 B tile into buffer `buf` -----
  auto stage = [&](int buf, int t) {
    const float4* ap = (const float4*)(apg + t * 32);
    unsigned int* asp = (unsigned int*)&As[buf][arow][acol];
#pragma unroll
    for (int i = 0; i < 4; ++i) {
      float4 tv = ap[i];
      asp[2 * i]     = pack2bf(tv.x, tv.y);
      asp[2 * i + 1] = pack2bf(tv.z, tv.w);
    }
    const float* bp = bpg + (size_t)t * 32 * N;
    float4 r0a = ((const float4*)(bp))[0],         r0b = ((const float4*)(bp))[1];
    float4 r1a = ((const float4*)(bp + N))[0],     r1b = ((const float4*)(bp + N))[1];
    float4 r2a = ((const float4*)(bp + 2 * N))[0], r2b = ((const float4*)(bp + 2 * N))[1];
    float4 r3a = ((const float4*)(bp + 3 * N))[0], r3b = ((const float4*)(bp + 3 * N))[1];
    float q0[8] = {r0a.x, r0a.y, r0a.z, r0a.w, r0b.x, r0b.y, r0b.z, r0b.w};
    float q1[8] = {r1a.x, r1a.y, r1a.z, r1a.w, r1b.x, r1b.y, r1b.z, r1b.w};
    float q2[8] = {r2a.x, r2a.y, r2a.z, r2a.w, r2b.x, r2b.y, r2b.z, r2b.w};
    float q3[8] = {r3a.x, r3a.y, r3a.z, r3a.w, r3b.x, r3b.y, r3b.z, r3b.w};
#pragma unroll
    for (int cc = 0; cc < 8; ++cc) {
      *(unsigned int*)&Bs[buf][bcol8 + cc][bk4]     = pack2bf(q0[cc], q1[cc]);
      *(unsigned int*)&Bs[buf][bcol8 + cc][bk4 + 2] = pack2bf(q2[cc], q3[cc]);
    }
  };

  stage(0, 0);
  __syncthreads();

  for (int t = 0; t < nk; ++t) {
    int cur = t & 1;
    if (t + 1 < nk) stage(cur ^ 1, t + 1);
    if (t + 2 < nk) {  // prefetch tile t+2 into cache (global_prefetch_b8)
      __builtin_prefetch(apg + (t + 2) * 32, 0, 0);
      __builtin_prefetch(bpg + (size_t)(t + 2) * 32 * N, 0, 0);
    }

    // fragment loads per ISA VGPR layouts
    BF16Frag af[4], bfr[4];
#pragma unroll
    for (int mi = 0; mi < 4; ++mi) {
      int row = waveM * 64 + mi * 16 + l16;
#pragma unroll
      for (int j = 0; j < 8; ++j) {
        int k = 2 * j + ((j >= 4) ? 8 : 0) + hi * 8;  // A 16x32 layout
        af[mi].u[j] = *(const unsigned int*)&As[cur][row][k];
      }
    }
#pragma unroll
    for (int ni = 0; ni < 4; ++ni) {
      int col = waveN * 64 + ni * 16 + l16;
#pragma unroll
      for (int j = 0; j < 8; ++j) {
        int k = 2 * j + ((j >= 4) ? 8 : 0) + hi * 8;  // B 32x16 layout
        bfr[ni].u[j] = *(const unsigned int*)&Bs[cur][col][k];
      }
    }
#pragma unroll
    for (int mi = 0; mi < 4; ++mi)
#pragma unroll
      for (int ni = 0; ni < 4; ++ni)
        acc[mi][ni].v = WMMA_BF16(af[mi].v, bfr[ni].v, acc[mi][ni].v);
    __syncthreads();
  }

  // --- epilogue (compile-time specialized: no per-element branches) ---
#pragma unroll
  for (int mi = 0; mi < 4; ++mi)
#pragma unroll
    for (int ni = 0; ni < 4; ++ni)
#pragma unroll
      for (int r = 0; r < 8; ++r) {
        int row = m0 + waveM * 64 + mi * 16 + hi * 8 + r;
        int col = n0 + waveN * 64 + ni * 16 + l16;
        float v = acc[mi][ni].f[r];
        if (HAS_BIAS) v += bias[col];
        if (ACT == 1)  // fast stable softplus: v_exp + v_log
          v = fmaxf(v, 0.f) + __logf(1.f + __expf(-fabsf(v)));
        size_t idx = (size_t)row * N + col;
        if (HAS_RES) v += res[idx];
        C[idx] = v;
      }
}

// ---------------------------------------------------------------------------
// Pack q / kv (f32) to bf16:  qb,kb = [B,H,S,64]; vt = [B,H,64,S] (transposed)
// ---------------------------------------------------------------------------
__global__ __launch_bounds__(256) void pack_qkv(
    const float* __restrict__ q, const float* __restrict__ kv,
    unsigned short* __restrict__ qb, unsigned short* __restrict__ kb,
    unsigned short* __restrict__ vt) {
  int i = blockIdx.x * 256 + threadIdx.x;  // 4M elements
  int d = i & 63, h = (i >> 6) & 15, s = (i >> 10) & 2047, b = i >> 21;
  size_t bh = (size_t)(b * NH + h);
  size_t rsd = ((size_t)(b * SEQ + s));
  qb[(bh * SEQ + s) * HD + d] = (unsigned short)f2bf(q[rsd * MD + h * HD + d]);
  const float* kvr = kv + rsd * (2 * MD);
  kb[(bh * SEQ + s) * HD + d] = (unsigned short)f2bf(kvr[h * HD + d]);
  vt[(bh * HD + d) * SEQ + s] = (unsigned short)f2bf(kvr[MD + h * HD + d]);
}

// ---------------------------------------------------------------------------
// Flash attention (ALiBi, causal). Grid (S/64, H, B), 128 thr = 4 waves,
// each wave owns 16 query rows. WMMA for QK^T and P@V; online softmax.
// All 8 B-fragments are preloaded before each WMMA burst so the
// global_load_b128 clause overlaps the matrix pipe (partial loadcnt waits).
// ---------------------------------------------------------------------------
__global__ __launch_bounds__(128) void fa_kernel(
    const unsigned short* __restrict__ qb, const unsigned short* __restrict__ kb,
    const unsigned short* __restrict__ vt, float* __restrict__ attn) {
  int qt = blockIdx.x, h = blockIdx.y, b = blockIdx.z;
  int wid = threadIdx.x >> 5, lane = threadIdx.x & 31;
  int hi = (lane >> 4) & 1, l16 = lane & 15;
  int qbase = qt * 64 + wid * 16;
  size_t bh = (size_t)(b * NH + h);
  const unsigned short* qrow = qb + bh * SEQ * HD;
  const unsigned short* krow = kb + bh * SEQ * HD;
  const unsigned short* vrow = vt + bh * HD * SEQ;
  float slope = exp2f(-0.5f * (float)(h + 1));

  __shared__ unsigned short P[4][16][64];  // per-wave P staging (bf16)

  // Q A-fragments (16 rows x 64 over two 32-chunks)
  BF16Frag qa[2];
  int qr = qbase + l16;
#pragma unroll
  for (int c = 0; c < 2; ++c)
#pragma unroll
    for (int j = 0; j < 8; ++j) {
      int k = c * 32 + 2 * j + ((j >= 4) ? 8 : 0) + hi * 8;
      qa[c].u[j] = *(const unsigned int*)&qrow[(size_t)qr * HD + k];
    }

  F32Frag ou[4];
  float m[8], lsum[8];
#pragma unroll
  for (int r = 0; r < 8; ++r) { m[r] = -1e30f; lsum[r] = 0.f; }
#pragma unroll
  for (int hs = 0; hs < 4; ++hs)
#pragma unroll
    for (int r = 0; r < 8; ++r) ou[hs].f[r] = 0.f;

  for (int kt = 0; kt <= qt; ++kt) {
    int kb0 = kt * 64;

    // ---- preload all K fragments for this tile (8 x b128-pair clause) ----
    BF16Frag bk[4][2];
#pragma unroll
    for (int sub = 0; sub < 4; ++sub) {
      int key = kb0 + sub * 16 + l16;
#pragma unroll
      for (int c = 0; c < 2; ++c)
#pragma unroll
        for (int j = 0; j < 8; ++j) {
          int d = c * 32 + 2 * j + ((j >= 4) ? 8 : 0) + hi * 8;
          bk[sub][c].u[j] = *(const unsigned int*)&krow[(size_t)key * HD + d];
        }
    }
    // ---- S = Q @ K^T (16 x 64): 8 back-to-back WMMAs ----
    F32Frag su[4];
#pragma unroll
    for (int sub = 0; sub < 4; ++sub) {
      F32Frag cz;
#pragma unroll
      for (int r = 0; r < 8; ++r) cz.f[r] = 0.f;
      cz.v = WMMA_BF16(qa[0].v, bk[sub][0].v, cz.v);
      cz.v = WMMA_BF16(qa[1].v, bk[sub][1].v, cz.v);
      su[sub] = cz;
    }
    // ---- scale + ALiBi + causal mask, row max ----
    float mt[8];
#pragma unroll
    for (int r = 0; r < 8; ++r) {
      int qg = qbase + hi * 8 + r;
      float rmax = -1e30f;
#pragma unroll
      for (int sub = 0; sub < 4; ++sub) {
        int kg = kb0 + sub * 16 + l16;
        float v = su[sub].f[r] * 0.125f +
                  ((kg <= qg) ? slope * (float)(kg - qg) : -9.0e7f);
        su[sub].f[r] = v;
        rmax = fmaxf(rmax, v);
      }
      rmax = fmaxf(rmax, __shfl_xor(rmax, 1));
      rmax = fmaxf(rmax, __shfl_xor(rmax, 2));
      rmax = fmaxf(rmax, __shfl_xor(rmax, 4));
      rmax = fmaxf(rmax, __shfl_xor(rmax, 8));
      mt[r] = rmax;
    }
    // ---- online softmax update ----
#pragma unroll
    for (int r = 0; r < 8; ++r) {
      float mnew = fmaxf(m[r], mt[r]);
      float corr = __expf(m[r] - mnew);
      float rs = 0.f;
#pragma unroll
      for (int sub = 0; sub < 4; ++sub) {
        float p = __expf(su[sub].f[r] - mnew);
        su[sub].f[r] = p;
        rs += p;
      }
      rs += __shfl_xor(rs, 1);
      rs += __shfl_xor(rs, 2);
      rs += __shfl_xor(rs, 4);
      rs += __shfl_xor(rs, 8);
      lsum[r] = lsum[r] * corr + rs;
      m[r] = mnew;
#pragma unroll
      for (int hs = 0; hs < 4; ++hs) ou[hs].f[r] *= corr;
    }
    // ---- C-layout -> A-layout via per-wave LDS (same-wave DS is in-order) ----
#pragma unroll
    for (int sub = 0; sub < 4; ++sub)
#pragma unroll
      for (int r = 0; r < 8; ++r)
        P[wid][hi * 8 + r][sub * 16 + l16] = (unsigned short)f2bf(su[sub].f[r]);
    BF16Frag pa[2];
#pragma unroll
    for (int c = 0; c < 2; ++c)
#pragma unroll
      for (int j = 0; j < 8; ++j) {
        int k = c * 32 + 2 * j + ((j >= 4) ? 8 : 0) + hi * 8;
        pa[c].u[j] = *(const unsigned int*)&P[wid][l16][k];
      }
    // ---- preload all V fragments, then O += P @ V (8 WMMAs) ----
    BF16Frag bv[4][2];
#pragma unroll
    for (int hs = 0; hs < 4; ++hs) {
      int dcol = hs * 16 + l16;
#pragma unroll
      for (int c = 0; c < 2; ++c)
#pragma unroll
        for (int j = 0; j < 8; ++j) {
          int key = kb0 + c * 32 + 2 * j + ((j >= 4) ? 8 : 0) + hi * 8;
          bv[hs][c].u[j] = *(const unsigned int*)&vrow[(size_t)dcol * SEQ + key];
        }
    }
#pragma unroll
    for (int hs = 0; hs < 4; ++hs) {
      ou[hs].v = WMMA_BF16(pa[0].v, bv[hs][0].v, ou[hs].v);
      ou[hs].v = WMMA_BF16(pa[1].v, bv[hs][1].v, ou[hs].v);
    }
  }

  // ---- normalize + store to [B,S,D] ----
#pragma unroll
  for (int hs = 0; hs < 4; ++hs)
#pragma unroll
    for (int r = 0; r < 8; ++r) {
      int qg  = qbase + hi * 8 + r;
      int col = h * HD + hs * 16 + l16;
      attn[((size_t)(b * SEQ) + qg) * MD + col] = ou[hs].f[r] / lsum[r];
    }
}

// ---------------------------------------------------------------------------
// Host orchestration
// ---------------------------------------------------------------------------
extern "C" void kernel_launch(void* const* d_in, const int* in_sizes, int n_in,
                              void* d_out, int out_size, void* d_ws, size_t ws_size,
                              hipStream_t stream) {
  (void)in_sizes; (void)n_in; (void)out_size; (void)ws_size;
  const float* tgt = (const float*)d_in[0];
  const float* fgm = (const float*)d_in[1];
  const float* fbt = (const float*)d_in[2];
  const float* wq  = (const float*)d_in[3];
  const float* bq  = (const float*)d_in[4];
  const float* wkv = (const float*)d_in[5];
  const float* bkv = (const float*)d_in[6];
  const float* wo  = (const float*)d_in[7];
  const float* bo  = (const float*)d_in[8];
  const float* w1  = (const float*)d_in[9];
  const float* b1  = (const float*)d_in[10];
  const float* w2  = (const float*)d_in[11];
  const float* b2  = (const float*)d_in[12];
  const float* ag  = (const float*)d_in[13];
  const float* ab  = (const float*)d_in[14];
  const float* fg  = (const float*)d_in[15];
  const float* fb  = (const float*)d_in[16];

  // Workspace layout (~120 MB):
  //   x (4M f32) | h (4M f32) | R1 (16M f32, reused: {q,kv,attn} then ffn1)
  //   | qb,kb,vt (3 x 4M bf16)
  float* x  = (float*)d_ws;
  float* h  = x + (size_t)ROWS * MD;
  float* r1 = h + (size_t)ROWS * MD;
  float* q    = r1;
  float* kv   = r1 + (size_t)ROWS * MD;  // 8M floats (k|v)
  float* attn = r1;                      // aliases q (q dead after pack)
  float* ffn1 = r1;                      // aliases everything (post-attn)
  unsigned short* qb = (unsigned short*)(r1 + (size_t)ROWS * FFN);
  unsigned short* kb = qb + (size_t)ROWS * MD;
  unsigned short* vt = kb + (size_t)ROWS * MD;

  hipMemcpyAsync(x, tgt, (size_t)ROWS * MD * sizeof(float),
                 hipMemcpyDeviceToDevice, stream);

  for (int i = 0; i < 8; ++i) {
    // attention block
    ln_kernel<<<ROWS, 256, 0, stream>>>(x, ag + i * MD, ab + i * MD, h);
    gemm_kernel<0, true, false><<<dim3(4, 32), 256, 0, stream>>>(
        h, wq + (size_t)i * MD * MD, bq + i * MD, nullptr, q, ROWS, MD, MD);
    gemm_kernel<0, true, false><<<dim3(8, 32), 256, 0, stream>>>(
        h, wkv + (size_t)i * MD * 2 * MD, bkv + i * 2 * MD, nullptr, kv,
        ROWS, 2 * MD, MD);
    pack_qkv<<<(ROWS * MD) / 256, 256, 0, stream>>>(q, kv, qb, kb, vt);
    fa_kernel<<<dim3(SEQ / 64, NH, BATCH), 128, 0, stream>>>(qb, kb, vt, attn);
    gemm_kernel<0, true, true><<<dim3(4, 32), 256, 0, stream>>>(
        attn, wo + (size_t)i * MD * MD, bo + i * MD, x, x, ROWS, MD, MD);
    // FFN block
    ln_kernel<<<ROWS, 256, 0, stream>>>(x, fg + i * MD, fb + i * MD, h);
    gemm_kernel<1, true, false><<<dim3(16, 32), 256, 0, stream>>>(
        h, w1 + (size_t)i * MD * FFN, b1 + i * FFN, nullptr, ffn1,
        ROWS, FFN, MD);
    gemm_kernel<0, true, true><<<dim3(4, 32), 256, 0, stream>>>(
        ffn1, w2 + (size_t)i * FFN * MD, b2 + i * MD, x, x, ROWS, MD, FFN);
  }
  ln_kernel<<<ROWS, 256, 0, stream>>>(x, fgm, fbt, (float*)d_out);
}
